// MSE_DQ_11613591569168
// MI455X (gfx1250) — matrix operand order repaired and checked by reading, not compiled, not run on hardware
//
#include <hip/hip_runtime.h>
#include <hip/hip_bf16.h>

// ---------------- CDNA5 WMMA types ----------------
typedef __bf16 bf16_t;
typedef bf16_t v16bf __attribute__((ext_vector_type(16)));
typedef float  v8f   __attribute__((ext_vector_type(8)));

// ---------------- problem constants ----------------
#define BP    512
#define MROWS 1024          // Bp*2
#define ZDIM  256
#define CCH   96
#define TT    128
#define CT    12288         // C*T
#define TGT_MSTRIDE 24576   // 192*128
#define NJ    24

// parents from the reference setup (compile-time constant -> full unroll, static indexing)
constexpr int PAR[NJ] = {0,0,0,0,1,2,3,4,5,6,7,8,9,9,9,12,13,14,16,17,18,19,20,21};

// ---------------- quaternion helpers ----------------
struct Q  { float w,x,y,z; };
struct V3 { float x,y,z; };

__device__ __forceinline__ Q qmul(Q a, Q b){
  return Q{ a.w*b.w - a.x*b.x - a.y*b.y - a.z*b.z,
            a.w*b.x + a.x*b.w + a.y*b.z - a.z*b.y,
            a.w*b.y - a.x*b.z + a.y*b.w + a.z*b.x,
            a.w*b.z + a.x*b.y - a.y*b.x + a.z*b.w };
}
__device__ __forceinline__ Q qconj(Q a){ return Q{a.w,-a.x,-a.y,-a.z}; }
__device__ __forceinline__ float qdot(Q a, Q b){ return a.w*b.w + a.x*b.x + a.y*b.y + a.z*b.z; }
__device__ __forceinline__ Q qscale(Q a, float s){ return Q{a.w*s, a.x*s, a.y*s, a.z*s}; }
__device__ __forceinline__ Q qadd(Q a, Q b){ return Q{a.w+b.w, a.x+b.x, a.y+b.y, a.z+b.z}; }
__device__ __forceinline__ Q qinv(Q a){ float s = 1.0f/qdot(a,a); return qscale(qconj(a), s); }
__device__ __forceinline__ V3 v3add(V3 a, V3 b){ return V3{a.x+b.x, a.y+b.y, a.z+b.z}; }
__device__ __forceinline__ V3 qrot(Q q, V3 v){
  Q vq{0.f, v.x, v.y, v.z};
  Q r = qmul(qmul(q, vq), qinv(q));
  return V3{r.x, r.y, r.z};
}
// adjoint of y = qinv(q) = conj(q)/|q|^2 given ybar
__device__ __forceinline__ Q qinv_adj(Q q, Q yb){
  float s  = qdot(q,q);
  float is = 1.0f/s, is2 = is*is;
  float d  = yb.w*q.w - yb.x*q.x - yb.y*q.y - yb.z*q.z;  // dot(yb, conj(q))
  return qadd(qscale(qconj(yb), is), qscale(q, -2.0f*d*is2));
}
// adjoint wrt q of r = quat_rotate(q, v) given rbar (v grad unused by reference)
__device__ __forceinline__ Q qrot_adj_q(Q q, V3 v, V3 rb){
  Q vq{0.f, v.x, v.y, v.z};
  Q t   = qmul(q, vq);
  Q iq  = qinv(q);
  Q rb4{0.f, rb.x, rb.y, rb.z};
  Q tb  = qmul(rb4, qconj(iq));     // through r4 = t (x) iq, first arg
  Q iqb = qmul(qconj(t), rb4);      // second arg
  return qadd(qmul(tb, qconj(vq)), qinv_adj(q, iqb));
}

// ---------------- block reduction (blockDim.x == 256) ----------------
__device__ __forceinline__ float block_reduce256(float v){
  __shared__ float s[256];
  int tid = threadIdx.x;
  s[tid] = v; __syncthreads();
  #pragma unroll
  for (int o = 128; o > 0; o >>= 1){
    if (tid < o) s[tid] += s[tid+o];
    __syncthreads();
  }
  float r = s[0];
  __syncthreads();
  return r;
}

// generic-pointer -> 32-bit LDS byte address (flat LDS addressing uses addr[31:0])
__device__ __forceinline__ unsigned lds_addr32(const void* p){
  return (unsigned)(unsigned long long)p;
}

// =====================================================================
// Kernel 1: qs = base + latent @ W_dec        (M=1024, K=256, N=12288)
// one wave per 16x16 tile; B tile staged LDS<-global via async b128
// (double buffered, ASYNCcnt-tracked), bf16 WMMA f32-acc, fused +base.
// =====================================================================
__global__ void __launch_bounds__(32)
gemm_decode_kernel(const float* __restrict__ lat,   // (1024,256)
                   const float* __restrict__ Wd,    // (256,12288)
                   const float* __restrict__ base,  // (1024,12288)
                   float* __restrict__ qs)          // (1024,12288)
{
  __shared__ float Btile[2][32 * 16];               // two 2KB K-step tiles

  const int tn   = blockIdx.x * 16;
  const int tm   = blockIdx.y * 16;
  const int lane = threadIdx.x;
  const int lrow = lane & 15;
  const int half = lane >> 4;

  // async-stage one 32(K) x 16(N) f32 tile of W_dec into LDS buffer `buf`
  // lane -> (row r, 16B chunk c4): 32 lanes cover 8 rows per issue, 4 issues.
  auto issue_tile = [&](int buf, int kb){
    const unsigned lbase = lds_addr32(&Btile[buf][0]);
    #pragma unroll
    for (int i = 0; i < 4; ++i){
      const int r  = i * 8 + (lane >> 2);
      const int c4 = (lane & 3) * 4;
      const float* g = Wd + (size_t)(kb + r) * CT + tn + c4;
      const unsigned l = lbase + (unsigned)((r * 16 + c4) * 4);
      asm volatile("global_load_async_to_lds_b128 %0, %1, off"
                   :: "v"(l), "v"(g) : "memory");
    }
  };

  issue_tile(0, 0);

  v8f acc = {};
  #pragma unroll
  for (int kk = 0; kk < 8; ++kk){
    const int kb = kk * 32;
    if (kk < 7) issue_tile((kk + 1) & 1, kb + 32);

    // ---- A fragment (direct global, per-lane contiguous b128s)
    v16bf a;
    const float* arow = lat + (size_t)(tm + lrow) * ZDIM + kb + half * 16;
    __builtin_prefetch(arow + 32, 0, 0);
    #pragma unroll
    for (int e = 0; e < 8; ++e) a[e]     = (bf16_t)arow[e];
    #pragma unroll
    for (int e = 0; e < 8; ++e) a[8 + e] = (bf16_t)arow[8 + e];

    // ---- wait for current B tile (4 outstanding belong to the next tile)
    if (kk < 7) asm volatile("s_wait_asynccnt 0x4" ::: "memory");
    else        asm volatile("s_wait_asynccnt 0x0" ::: "memory");

    // ---- B fragment from LDS: lane N = lrow; lanes 0-15 K 0-15, 16-31 K 16-31
    const float* bb = &Btile[kk & 1][0];
    v16bf b;
    #pragma unroll
    for (int e = 0; e < 16; ++e)
      b[e] = (bf16_t)bb[(half * 16 + e) * 16 + lrow];

    acc = __builtin_amdgcn_wmma_f32_16x16x32_bf16(false, a, false, b,
                                                  (short)0, acc, false, false);
  }

  // ---- epilogue: D layout: VGPR r -> M = half*8 + r, N = lrow ; fuse +base
  #pragma unroll
  for (int r = 0; r < 8; ++r){
    size_t idx = (size_t)(tm + half * 8 + r) * CT + (tn + lrow);
    qs[idx] = acc[r] + base[idx];
  }
}

// =====================================================================
// Kernel 2: grad_lat[b,z] = sum_ct gqs[b,ct] * W_dec[z,ct]
//           (M=512, K=12288, N=256) -- both operands per-lane contiguous
// =====================================================================
__global__ void __launch_bounds__(32)
gemm_gradlat_kernel(const float* __restrict__ gqs,  // (512,12288)
                    const float* __restrict__ Wd,   // (256,12288)
                    float* __restrict__ glat)       // (512,256)
{
  const int tn   = blockIdx.x * 16;   // over Z
  const int tm   = blockIdx.y * 16;   // over 512
  const int lane = threadIdx.x;
  const int lrow = lane & 15;
  const int half = lane >> 4;

  v8f acc = {};
  for (int kb = 0; kb < CT; kb += 32){
    v16bf a;
    const float* arow = gqs + (size_t)(tm + lrow) * CT + kb + half * 16;
    __builtin_prefetch(arow + 32, 0, 0);
    #pragma unroll
    for (int e = 0; e < 8; ++e) a[e]     = (bf16_t)arow[e];
    #pragma unroll
    for (int e = 0; e < 8; ++e) a[8 + e] = (bf16_t)arow[8 + e];

    // B[k][n] = Wd[n][k]  -> per-lane contiguous along k
    v16bf b;
    const float* brow = Wd + (size_t)(tn + lrow) * CT + kb + half * 16;
    __builtin_prefetch(brow + 32, 0, 0);
    #pragma unroll
    for (int e = 0; e < 16; ++e) b[e] = (bf16_t)brow[e];

    acc = __builtin_amdgcn_wmma_f32_16x16x32_bf16(false, a, false, b,
                                                  (short)0, acc, false, false);
  }
  #pragma unroll
  for (int r = 0; r < 8; ++r)
    glat[(size_t)(tm + half * 8 + r) * ZDIM + (tn + lrow)] = acc[r];
}

// =====================================================================
// Kernel 3: root + joints MSE partial sums (over qs vs target channels)
// =====================================================================
__global__ void rj_kernel(const float* __restrict__ qs,
                          const float* __restrict__ tgt,
                          float* __restrict__ acc)
{
  const long long n = (long long)MROWS * CCH * TT;
  float sr = 0.f, sj = 0.f;
  for (long long i = blockIdx.x * blockDim.x + threadIdx.x; i < n;
       i += (long long)gridDim.x * blockDim.x){
    int m = (int)(i / CT);
    int r = (int)(i - (long long)m * CT);
    int c = r >> 7;
    int t = r & 127;
    float a  = qs[i];
    float bt = tgt[(size_t)m * TGT_MSTRIDE + (size_t)(((c >> 2) << 3) + (c & 3)) * TT + t];
    float d  = (a - bt) * (a - bt);
    if (c < 4) sr += d; else sj += d;
  }
  float r0 = block_reduce256(sr);
  float r1 = block_reduce256(sj);
  if (threadIdx.x == 0){ atomicAdd(acc + 1, r0); atomicAdd(acc + 5, r1); }
}

// =====================================================================
// Kernel 4: generic squared-difference reduction
// =====================================================================
__global__ void sqdiff_kernel(const float* __restrict__ a,
                              const float* __restrict__ b,
                              long long n, float* __restrict__ acc)
{
  float s = 0.f;
  for (long long i = blockIdx.x * blockDim.x + threadIdx.x; i < n;
       i += (long long)gridDim.x * blockDim.x){
    float d = a[i] - b[i];
    s += d * d;
  }
  float r = block_reduce256(s);
  if (threadIdx.x == 0) atomicAdd(acc, r);
}

// =====================================================================
// Kernel 5: KLD partial sum:  sum(1 + logvar - mu^2 - exp(logvar))
// =====================================================================
__global__ void kld_kernel(const float* __restrict__ mu,
                           const float* __restrict__ lv,
                           float* __restrict__ acc)
{
  const long long n = (long long)MROWS * ZDIM;
  float s = 0.f;
  for (long long i = blockIdx.x * blockDim.x + threadIdx.x; i < n;
       i += (long long)gridDim.x * blockDim.x){
    float m = mu[i], l = lv[i];
    s += 1.0f + l - m * m - __expf(l);
  }
  float r = block_reduce256(s);
  if (threadIdx.x == 0) atomicAdd(acc + 0, r);
}

// =====================================================================
// Kernel 6: forward FK. one thread per (m,t) chain; parents hardcoded.
// =====================================================================
__global__ void fk_forward_kernel(const float* __restrict__ src,
                                  const float* __restrict__ meanq,
                                  const float* __restrict__ stdq,
                                  const float* __restrict__ offsets,
                                  float* __restrict__ pos,
                                  int jstride, int mstride)
{
  int gid = blockIdx.x * blockDim.x + threadIdx.x;
  if (gid >= MROWS * TT) return;
  int m = gid >> 7, t = gid & 127;
  int b = m >> 1;

  // scaled quats, root overridden with identity
  Q q[NJ];
  q[0] = Q{1.f, 0.f, 0.f, 0.f};
  #pragma unroll
  for (int j = 1; j < NJ; ++j){
    float comp[4];
    #pragma unroll
    for (int w = 0; w < 4; ++w){
      int c = 4 * j + w;
      comp[w] = src[(size_t)m * mstride + (size_t)(jstride * j + w) * TT + t] * stdq[c] + meanq[c];
    }
    q[j] = Q{comp[0], comp[1], comp[2], comp[3]};
  }

  // from_root_quat: keep j<4, else local = inv(q[p]) * q[j]
  Q loc[NJ];
  #pragma unroll
  for (int j = 0; j < NJ; ++j)
    loc[j] = (j < 4) ? q[j] : qmul(qinv(q[PAR[j]]), q[j]);

  // fk
  Q  grot[NJ];
  V3 gpos[NJ];
  grot[0] = loc[0];
  gpos[0] = V3{0.f, 0.f, 0.f};
  #pragma unroll
  for (int j = 1; j < NJ; ++j){
    int p = PAR[j];
    grot[j] = qmul(grot[p], loc[j]);
    V3 off = V3{ offsets[((size_t)b * NJ + j) * 3 + 0],
                 offsets[((size_t)b * NJ + j) * 3 + 1],
                 offsets[((size_t)b * NJ + j) * 3 + 2] };
    gpos[j] = v3add(gpos[p], qrot(grot[p], off));
  }

  #pragma unroll
  for (int j = 0; j < NJ; ++j){
    size_t o = ((size_t)gid * NJ + j) * 3;
    pos[o + 0] = gpos[j].x;
    pos[o + 1] = gpos[j].y;
    pos[o + 2] = gpos[j].z;
  }
}

// =====================================================================
// Kernel 7: backward FK for k=0 chains -> gqs (512,96,128)
// dL/dpos[b,0,t,j] = 2*(pos[b,0]-pos[b,1]); exact reverse-mode adjoints.
// =====================================================================
__global__ void fk_backward_kernel(const float* __restrict__ qs,
                                   const float* __restrict__ meanq,
                                   const float* __restrict__ stdq,
                                   const float* __restrict__ offsets,
                                   const float* __restrict__ pos,
                                   float* __restrict__ gqs)
{
  int gid = blockIdx.x * blockDim.x + threadIdx.x;
  if (gid >= BP * TT) return;
  int b = gid >> 7, t = gid & 127;
  int m0 = 2 * b;

  // ---- recompute forward for row m0
  Q q[NJ];
  q[0] = Q{1.f, 0.f, 0.f, 0.f};
  #pragma unroll
  for (int j = 1; j < NJ; ++j){
    float comp[4];
    #pragma unroll
    for (int w = 0; w < 4; ++w){
      int c = 4 * j + w;
      comp[w] = qs[(size_t)m0 * CT + (size_t)c * TT + t] * stdq[c] + meanq[c];
    }
    q[j] = Q{comp[0], comp[1], comp[2], comp[3]};
  }
  V3 off[NJ];
  #pragma unroll
  for (int j = 0; j < NJ; ++j)
    off[j] = V3{ offsets[((size_t)b * NJ + j) * 3 + 0],
                 offsets[((size_t)b * NJ + j) * 3 + 1],
                 offsets[((size_t)b * NJ + j) * 3 + 2] };
  Q loc[NJ];
  #pragma unroll
  for (int j = 0; j < NJ; ++j)
    loc[j] = (j < 4) ? q[j] : qmul(qinv(q[PAR[j]]), q[j]);
  Q grot[NJ];
  grot[0] = loc[0];
  #pragma unroll
  for (int j = 1; j < NJ; ++j) grot[j] = qmul(grot[PAR[j]], loc[j]);

  // ---- seed position grads: 2*(pos_k0 - pos_k1)
  V3 gpb[NJ];
  #pragma unroll
  for (int j = 0; j < NJ; ++j){
    size_t o0 = (((size_t)m0       * TT + t) * NJ + j) * 3;
    size_t o1 = (((size_t)(m0 + 1) * TT + t) * NJ + j) * 3;
    gpb[j] = V3{ 2.f * (pos[o0 + 0] - pos[o1 + 0]),
                 2.f * (pos[o0 + 1] - pos[o1 + 1]),
                 2.f * (pos[o0 + 2] - pos[o1 + 2]) };
  }

  // ---- reverse sweep: grot-bar / local-bar (children have larger index)
  Q grb[NJ];
  #pragma unroll
  for (int j = 0; j < NJ; ++j) grb[j] = Q{0.f, 0.f, 0.f, 0.f};
  Q locb[NJ];
  #pragma unroll
  for (int j = NJ - 1; j >= 1; --j){
    int p = PAR[j];
    // gpos[j] = gpos[p] + qrot(grot[p], off[j])
    grb[p] = qadd(grb[p], qrot_adj_q(grot[p], off[j], gpb[j]));
    gpb[p] = v3add(gpb[p], gpb[j]);
    // grot[j] = grot[p] (x) loc[j]
    grb[p]  = qadd(grb[p], qmul(grb[j], qconj(loc[j])));
    locb[j] = qmul(qconj(grot[p]), grb[j]);
  }
  locb[0] = grb[0];

  // ---- through from_root_quat
  Q qb[NJ];
  #pragma unroll
  for (int j = 0; j < NJ; ++j) qb[j] = Q{0.f, 0.f, 0.f, 0.f};
  #pragma unroll
  for (int j = 0; j < NJ; ++j){
    if (j < 4){
      qb[j] = qadd(qb[j], locb[j]);
    } else {
      int p = PAR[j];
      Q iq = qinv(q[p]);
      qb[j] = qadd(qb[j], qmul(qconj(iq), locb[j]));       // second-arg adjoint
      Q ib  = qmul(locb[j], qconj(q[j]));                  // first-arg adjoint -> iq-bar
      qb[p] = qadd(qb[p], qinv_adj(q[p], ib));
    }
  }

  // ---- through scaling; root channels overwritten -> zero grad
  size_t rb = (size_t)b * CT;
  #pragma unroll
  for (int w = 0; w < 4; ++w) gqs[rb + (size_t)w * TT + t] = 0.f;
  #pragma unroll
  for (int j = 1; j < NJ; ++j){
    float comp[4] = { qb[j].w, qb[j].x, qb[j].y, qb[j].z };
    #pragma unroll
    for (int w = 0; w < 4; ++w){
      int c = 4 * j + w;
      gqs[rb + (size_t)c * TT + t] = comp[w] * stdq[c];
    }
  }
}

// =====================================================================
// Kernel 8: consistency loss  sum((lat[2b] - glat[b] - lat[2b+1])^2)
// =====================================================================
__global__ void cons_kernel(const float* __restrict__ lat,
                            const float* __restrict__ glat,
                            float* __restrict__ acc)
{
  const long long n = (long long)BP * ZDIM;
  float s = 0.f;
  for (long long i = blockIdx.x * blockDim.x + threadIdx.x; i < n;
       i += (long long)gridDim.x * blockDim.x){
    int b = (int)(i >> 8), z = (int)(i & 255);
    float v = lat[(size_t)(2 * b) * ZDIM + z] - glat[i] - lat[(size_t)(2 * b + 1) * ZDIM + z];
    s += v * v;
  }
  float r = block_reduce256(s);
  if (threadIdx.x == 0) atomicAdd(acc + 3, r);
}

// =====================================================================
// Kernel 9: finalize 6 outputs
// =====================================================================
__global__ void finalize_kernel(const float* __restrict__ acc, float* __restrict__ out)
{
  if (threadIdx.x == 0 && blockIdx.x == 0){
    out[0] = 0.001f * (-0.5f * acc[0] / 1024.0f);          // L_KLD * loss_kld
    out[1] = acc[1] / 524288.0f;                            // L_ROOT * loss_root
    out[2] = acc[2] / 393216.0f;                            // L_DISP * loss_disp
    out[3] = 0.1f  * (acc[3] / 131072.0f);                  // L_CONS * loss_cons
    out[4] = 0.1f  * (acc[4] / 9437184.0f);                 // L_FK   * loss_fk
    out[5] = acc[5] / 12058624.0f;                          // loss_joints
  }
}

// =====================================================================
extern "C" void kernel_launch(void* const* d_in, const int* in_sizes, int n_in,
                              void* d_out, int out_size, void* d_ws, size_t ws_size,
                              hipStream_t stream)
{
  const float* base   = (const float*)d_in[0];   // (512,2,96,128)
  const float* tgt    = (const float*)d_in[1];   // (512,2,192,128)
  const float* idisp  = (const float*)d_in[2];   // (512,2,3,128)
  const float* tdisp  = (const float*)d_in[3];   // (512,2,3,128)
  const float* mu     = (const float*)d_in[4];   // (1024,256)
  const float* logvar = (const float*)d_in[5];   // (1024,256)
  const float* lat    = (const float*)d_in[6];   // (1024,256)
  const float* Wd     = (const float*)d_in[7];   // (256,12288)
  const float* meanq  = (const float*)d_in[8];   // (96,1)
  const float* stdq   = (const float*)d_in[9];   // (96,1)
  const float* offs   = (const float*)d_in[10];  // (512,24,3)
  // d_in[11] = parents (hardcoded at compile time)

  float* out = (float*)d_out;

  // ---- workspace layout ----
  float* acc   = (float*)d_ws;                                   // 8 floats (64B hdr)
  float* qs    = (float*)((char*)d_ws + 256);                    // 1024*12288
  float* pos_q = qs    + (size_t)MROWS * CT;                     // 1024*128*24*3
  float* pos_t = pos_q + (size_t)MROWS * TT * NJ * 3;
  float* gqs   = pos_t + (size_t)MROWS * TT * NJ * 3;            // 512*12288
  float* glat  = gqs   + (size_t)BP * CT;                        // 512*256

  hipMemsetAsync(acc, 0, 64, stream);

  // 1) decode GEMM (WMMA + async global->LDS staging of B tiles)
  dim3 g1(CT / 16, MROWS / 16);
  gemm_decode_kernel<<<g1, 32, 0, stream>>>(lat, Wd, base, qs);

  // 2) elementwise losses
  rj_kernel<<<2048, 256, 0, stream>>>(qs, tgt, acc);
  sqdiff_kernel<<<512, 256, 0, stream>>>(idisp, tdisp, (long long)BP * 2 * 3 * TT, acc + 2);
  kld_kernel<<<512, 256, 0, stream>>>(mu, logvar, acc);

  // 3) FK for decoded and target motion
  fk_forward_kernel<<<(MROWS * TT) / 128, 128, 0, stream>>>(qs,  meanq, stdq, offs, pos_q, 4, CT);
  fk_forward_kernel<<<(MROWS * TT) / 128, 128, 0, stream>>>(tgt, meanq, stdq, offs, pos_t, 8, TGT_MSTRIDE);
  sqdiff_kernel<<<2048, 256, 0, stream>>>(pos_q, pos_t, (long long)MROWS * TT * NJ * 3, acc + 4);

  // 4) backward FK (k=0 rows) -> gqs
  fk_backward_kernel<<<(BP * TT) / 128, 128, 0, stream>>>(qs, meanq, stdq, offs, pos_q, gqs);

  // 5) grad_lat GEMM (WMMA)
  dim3 g2(ZDIM / 16, BP / 16);
  gemm_gradlat_kernel<<<g2, 32, 0, stream>>>(gqs, Wd, glat);

  // 6) consistency + finalize
  cons_kernel<<<256, 256, 0, stream>>>(lat, glat, acc);
  finalize_kernel<<<1, 32, 0, stream>>>(acc, out);
}
